// SwinDecoderBlock_87033217286285
// MI455X (gfx1250) — compile-verified, matrix operand order checked by reading, not run on hardware
//
#include <hip/hip_runtime.h>
#include <hip/hip_bf16.h>
#include <cstddef>

typedef _Float16 f16;
typedef __attribute__((ext_vector_type(16))) _Float16 v16h;
typedef __attribute__((ext_vector_type(8)))  _Float16 v8h;
typedef __attribute__((ext_vector_type(8)))  float    v8f;

// ---------------- problem constants ----------------
#define BB     2
#define HI     60
#define WI     60
#define TT     4
#define WS     8
#define SHIFT  4
#define CC     256
#define NH     8
#define HD     32          // CC/NH
#define HID    1024
#define PP     (HI*WI)                 // 3600
#define NTOK   (BB*TT*PP)              // 28800 rows
#define BM_IMG (BB*TT)                 // 8 images
#define NWIN   (BM_IMG*64)             // 512 windows
#define LL     (WS*WS)                 // 64
#define NWL    (NWIN*LL)               // 32768 window rows
#define SCALE_F 0.1767766953f          // 1/sqrt(32)

// ---------------- fragment helpers ----------------
union V16U { v16h v; v8h h[2]; };

// 16-bit A-matrix (or B==K^T) fragment: lane holds row (lane&15) of a 16x32
// tile stored row-major with leading dimension ld. Halves 0..7 = K kb..kb+7,
// halves 8..15 = K kb+16..kb+23, kb = (lane>=16)?8:0.
__device__ __forceinline__ v16h load_frag_rm(const f16* base, int ld, int lane) {
    int r  = lane & 15;
    int kb = (lane >> 4) ? 8 : 0;
    const f16* p = base + (size_t)r * ld + kb;
    V16U u;
    u.h[0] = *(const v8h*)(p);
    u.h[1] = *(const v8h*)(p + 16);
    return u.v;
}

// B fragment from a K-major matrix B[k][n] with leading dimension ld:
// lane holds column n = lane&15, same K half-pattern (strided element loads).
__device__ __forceinline__ v16h load_frag_kmaj(const f16* base, int ld, int lane) {
    int n  = lane & 15;
    int kb = (lane >> 4) ? 8 : 0;
    V16U u;
#pragma unroll
    for (int t = 0; t < 8; ++t) u.v[t]     = base[(size_t)(kb + t) * ld + n];
#pragma unroll
    for (int t = 0; t < 8; ++t) u.v[8 + t] = base[(size_t)(kb + 16 + t) * ld + n];
    return u.v;
}

__device__ __forceinline__ v8f wmma16(v16h a, v16h b, v8f c) {
    return __builtin_amdgcn_wmma_f32_16x16x32_f16(false, a, false, b, (short)0, c,
                                                  false, false);
}

// gfx1250 async memory->LDS copy (16B per lane), tracked by ASYNCcnt.
// vdst VGPR holds the 32-bit LDS byte offset (low 32 bits of flat shared addr).
__device__ __forceinline__ void async_load_b128(const f16* gsrc, f16* ldst) {
    unsigned lds_off = (unsigned)(unsigned long long)(const void*)ldst;
    asm volatile("global_load_async_to_lds_b128 %0, %1, off"
                 :: "v"(lds_off), "v"(gsrc)
                 : "memory");
}
__device__ __forceinline__ void wait_async0() {
    asm volatile("s_wait_asynccnt 0x0" ::: "memory");
}

// ---------------- elementwise kernels ----------------
__global__ void f32_to_f16_kernel(const float* __restrict__ in, f16* __restrict__ out, int n) {
    int i = blockIdx.x * 256 + threadIdx.x;
    if (i < n) out[i] = (f16)in[i];
}

__global__ void copy_f32_kernel(const float* __restrict__ in, float* __restrict__ out, int n) {
    int i = blockIdx.x * 256 + threadIdx.x;
    if (i < n) out[i] = in[i];
}

__global__ void final_add_kernel(const float* __restrict__ X, const float* __restrict__ Hm,
                                 const float* __restrict__ g, float* __restrict__ out, int n) {
    int i = blockIdx.x * 256 + threadIdx.x;
    if (i < n) out[i] = X[i] + g[0] * Hm[i];
}

// LayerNorm over rows of 256 fp32 -> f16. One wave per row (wave32).
__global__ void __launch_bounds__(256)
ln_f16_kernel(const float* __restrict__ x, const float* __restrict__ w,
              const float* __restrict__ b, f16* __restrict__ out, int rows) {
    int wave = threadIdx.x >> 5, lane = threadIdx.x & 31;
    int row = blockIdx.x * 8 + wave;
    if (row >= rows) return;
    const float* xp = x + (size_t)row * CC + lane * 8;
    float v[8];
    float s = 0.f, sq = 0.f;
#pragma unroll
    for (int j = 0; j < 8; ++j) { v[j] = xp[j]; s += v[j]; sq += v[j] * v[j]; }
#pragma unroll
    for (int d = 16; d >= 1; d >>= 1) {
        s  += __shfl_xor(s, d, 32);
        sq += __shfl_xor(sq, d, 32);
    }
    float mean = s * (1.0f / CC);
    float var  = sq * (1.0f / CC) - mean * mean;
    float rstd = rsqrtf(var + 1e-5f);
    f16* op = out + (size_t)row * CC + lane * 8;
    const float* wp = w + lane * 8;
    const float* bp = b + lane * 8;
#pragma unroll
    for (int j = 0; j < 8; ++j) op[j] = (f16)((v[j] - mean) * rstd * wp[j] + bp[j]);
}

// roll(-4,-4) + pad + 8x8 window partition. One block per (window,row-in-window).
__global__ void __launch_bounds__(256)
gather_windows_kernel(const f16* __restrict__ src, f16* __restrict__ dst) {
    int wl = blockIdx.x;           // 0..NWL-1
    int w = wl >> 6, l = wl & 63;
    int m = w >> 6;                // image
    int widx = w & 63;
    int pr = ((widx >> 3) << 3) + (l >> 3);
    int pc = ((widx & 7) << 3) + (l & 7);
    int ch = threadIdx.x;
    f16 v = (f16)0.0f;
    if (pr < HI && pc < WI) {
        int sr = (pr + SHIFT) % HI;
        int sc = (pc + SHIFT) % WI;
        v = src[(((size_t)m * HI + sr) * WI + sc) * CC + ch];
    }
    dst[(size_t)wl * CC + ch] = v;
}

// inverse partition + crop + roll(+4,+4) + residual: X += gamma * Y
__global__ void __launch_bounds__(256)
scatter_residual_kernel(float* __restrict__ X, const float* __restrict__ YW,
                        const float* __restrict__ gamma) {
    int row = blockIdx.x;          // 0..NTOK-1
    int ch = threadIdx.x;
    int m = row / PP;
    int t = row % PP;
    int ri = t / WI, ci = t % WI;
    int pr = (ri + HI - SHIFT) % HI;
    int pc = (ci + WI - SHIFT) % WI;
    int w = m * 64 + ((pr >> 3) << 3) + (pc >> 3);
    int l = ((pr & 7) << 3) + (pc & 7);
    X[(size_t)row * CC + ch] += gamma[0] * YW[((size_t)w * 64 + l) * CC + ch];
}

// ---------------- WMMA GEMM:  out = act(A[MxK] @ W[NxK]^T + bias) ----------------
// block = 256 threads (8 waves). block tile 128(M) x 64(N). wave -> 16-row strip.
// W tile staged in LDS [n][k] via gfx1250 async-to-LDS copies, double-buffered.
__global__ void __launch_bounds__(256)
gemm_wmma_kernel(const f16* __restrict__ A, const f16* __restrict__ W,
                 const float* __restrict__ bias, f16* __restrict__ outh,
                 float* __restrict__ outf, int M, int N, int K, int act) {
    __shared__ __align__(16) f16 Bs[2][64 * 32];   // ping-pong [n][k] tiles of W
    int wave = threadIdx.x >> 5, lane = threadIdx.x & 31;
    int n0 = blockIdx.x * 64;
    int m0 = blockIdx.y * 128;
    int rowA = m0 + wave * 16;

    // this thread's 16B slice of the cooperative W-tile copy
    int n  = threadIdx.x >> 2;         // 0..63
    int kk = (threadIdx.x & 3) * 8;    // 0,8,16,24
    const f16* wsrc = W + (size_t)(n0 + n) * K + kk;
    f16* ldst[2] = { &Bs[0][n * 32 + kk], &Bs[1][n * 32 + kk] };

    // preload tile 0 (async, no VGPR staging)
    async_load_b128(wsrc, ldst[0]);
    wait_async0();
    __syncthreads();

    v8f acc[4];
    v8f vzero = {};
#pragma unroll
    for (int t = 0; t < 4; ++t) acc[t] = vzero;

    int buf = 0;
    for (int k0 = 0; k0 < K; k0 += 32) {
        bool more = (k0 + 32 < K);
        if (more)   // overlap next tile's fill with this tile's WMMAs
            async_load_b128(wsrc + k0 + 32, ldst[buf ^ 1]);

        v16h a = load_frag_rm(A + (size_t)rowA * K + k0, K, lane);
        if (more)
            __builtin_prefetch(A + (size_t)rowA * K + k0 + 32, 0, 0);
        const f16* Bcur = &Bs[buf][0];
#pragma unroll
        for (int tj = 0; tj < 4; ++tj) {
            v16h b = load_frag_rm(Bcur + (tj * 16) * 32, 32, lane);
            acc[tj] = wmma16(a, b, acc[tj]);
        }
        if (more) {
            wait_async0();
            __syncthreads();
            buf ^= 1;
        }
    }

    int rbase = rowA + ((lane >> 4) ? 8 : 0);
    int cb = lane & 15;
#pragma unroll
    for (int tj = 0; tj < 4; ++tj) {
        int col = n0 + tj * 16 + cb;
        float bv = bias ? bias[col] : 0.0f;
#pragma unroll
        for (int i = 0; i < 8; ++i) {
            int r = rbase + i;
            if (r < M) {
                float v = acc[tj][i] + bv;
                if (act) v = 0.5f * v * (1.0f + erff(v * 0.70710678f));  // exact gelu
                size_t idx = (size_t)r * N + col;
                if (outh) outh[idx] = (f16)v;
                else      outf[idx] = v;
            }
        }
    }
}

// ---------------- fused window attention ----------------
// grid = (NWIN, 2), block = 128 (4 waves); head = blockIdx.y*4 + wave.
__global__ void __launch_bounds__(128)
win_attn_kernel(const f16* __restrict__ QP, const f16* __restrict__ KP,
                const f16* __restrict__ VP, const float* __restrict__ bias_table,
                f16* __restrict__ OW) {
    __shared__ __align__(16) f16 Pb[4 * LL * LL];   // 32 KB: P per wave
    int wave = threadIdx.x >> 5, lane = threadIdx.x & 31;
    int w = blockIdx.x;
    int h = blockIdx.y * 4 + wave;
    int widx = w & 63;
    int wh = widx >> 3, ww = widx & 7;

    const f16* qb = QP + ((size_t)w * LL) * CC + h * HD;
    const f16* kb = KP + ((size_t)w * LL) * CC + h * HD;
    const f16* vb = VP + ((size_t)w * LL) * CC + h * HD;
    f16* P = Pb + wave * LL * LL;

    // K fragments: B = K^T, so load with the row-major A-style loader.
    v16h kf[4];
#pragma unroll
    for (int tj = 0; tj < 4; ++tj)
        kf[tj] = load_frag_rm(kb + (size_t)(tj * 16) * CC, CC, lane);

    int half16 = (lane >> 4) * 8;
    int cb = lane & 15;
    v8f vzero = {};

    // ---- S = softmax(Q K^T * scale + bias + mask), streamed by 16-row tiles
    for (int ti = 0; ti < 4; ++ti) {
        v16h qf = load_frag_rm(qb + (size_t)(ti * 16) * CC, CC, lane);
        v8f s[4];
#pragma unroll
        for (int tj = 0; tj < 4; ++tj) s[tj] = wmma16(qf, kf[tj], vzero);

#pragma unroll
        for (int i = 0; i < 8; ++i) {
            int qi = ti * 16 + half16 + i;          // query index in window
            int ri = qi >> 3, ci = qi & 7;
            float m = -3.0e38f;
#pragma unroll
            for (int tj = 0; tj < 4; ++tj) {
                int j  = tj * 16 + cb;              // key index
                int rj = j >> 3, cj = j & 7;
                int rel = (ri - rj + 7) * 15 + (ci - cj + 7);
                float sv = s[tj][i] * SCALE_F + bias_table[rel * NH + h];
                bool masked = ((wh * 8 + rj) >= HI) || ((ww * 8 + cj) >= WI);
                sv = masked ? -3.0e38f : sv;
                s[tj][i] = sv;
                m = fmaxf(m, sv);
            }
#pragma unroll
            for (int d = 8; d >= 1; d >>= 1) m = fmaxf(m, __shfl_xor(m, d, 32));
            float sum = 0.f;
#pragma unroll
            for (int tj = 0; tj < 4; ++tj) {
                float p = __expf(s[tj][i] - m);
                s[tj][i] = p;
                sum += p;
            }
#pragma unroll
            for (int d = 8; d >= 1; d >>= 1) sum += __shfl_xor(sum, d, 32);
            float rinv = 1.0f / sum;
#pragma unroll
            for (int tj = 0; tj < 4; ++tj) {
                int j = tj * 16 + cb;
                P[(size_t)qi * LL + j] = (f16)(s[tj][i] * rinv);
            }
        }
    }
    __syncthreads();

    // ---- O = P @ V   (K=64 -> two K=32 chunks; N=HD=32 -> two N tiles)
    v16h vf[2][2];
#pragma unroll
    for (int kc = 0; kc < 2; ++kc)
#pragma unroll
        for (int tn = 0; tn < 2; ++tn)
            vf[kc][tn] = load_frag_kmaj(vb + (size_t)(kc * 32) * CC + tn * 16, CC, lane);

    for (int ti = 0; ti < 4; ++ti) {
        v16h p0 = load_frag_rm(P + (size_t)(ti * 16) * LL, LL, lane);
        v16h p1 = load_frag_rm(P + (size_t)(ti * 16) * LL + 32, LL, lane);
#pragma unroll
        for (int tn = 0; tn < 2; ++tn) {
            v8f o = wmma16(p0, vf[0][tn], vzero);
            o = wmma16(p1, vf[1][tn], o);
#pragma unroll
            for (int i = 0; i < 8; ++i) {
                int qi = ti * 16 + half16 + i;
                int d  = tn * 16 + cb;
                OW[((size_t)w * LL + qi) * CC + h * HD + d] = (f16)o[i];
            }
        }
    }
}

// ---------------- host side ----------------
static inline void* carve(char*& p, size_t bytes) {
    void* r = p;
    p += (bytes + 255) & ~(size_t)255;
    return r;
}

extern "C" void kernel_launch(void* const* d_in, const int* in_sizes, int n_in,
                              void* d_out, int out_size, void* d_ws, size_t ws_size,
                              hipStream_t stream) {
    (void)in_sizes; (void)n_in; (void)out_size; (void)ws_size;
    const float* x_in    = (const float*)d_in[0];
    const float* ctx_in  = (const float*)d_in[1];
    const float* ln1_w = (const float*)d_in[2],  *ln1_b = (const float*)d_in[3];
    const float* ln2_w = (const float*)d_in[4],  *ln2_b = (const float*)d_in[5];
    const float* ln3_w = (const float*)d_in[6],  *ln3_b = (const float*)d_in[7];
    const float* lnc_w = (const float*)d_in[8],  *lnc_b = (const float*)d_in[9];
    const float* sa_q_w = (const float*)d_in[10], *sa_q_b = (const float*)d_in[11];
    const float* sa_k_w = (const float*)d_in[12], *sa_k_b = (const float*)d_in[13];
    const float* sa_v_w = (const float*)d_in[14], *sa_v_b = (const float*)d_in[15];
    const float* sa_o_w = (const float*)d_in[16], *sa_o_b = (const float*)d_in[17];
    const float* sa_bias = (const float*)d_in[18];
    const float* ca_q_w = (const float*)d_in[19], *ca_q_b = (const float*)d_in[20];
    const float* ca_k_w = (const float*)d_in[21], *ca_k_b = (const float*)d_in[22];
    const float* ca_v_w = (const float*)d_in[23], *ca_v_b = (const float*)d_in[24];
    const float* ca_o_w = (const float*)d_in[25], *ca_o_b = (const float*)d_in[26];
    const float* ca_bias = (const float*)d_in[27];
    const float* gamma1 = (const float*)d_in[28];
    const float* gamma2 = (const float*)d_in[29];
    const float* gamma_mlp = (const float*)d_in[30];
    const float* mlp_w1 = (const float*)d_in[31], *mlp_b1 = (const float*)d_in[32];
    const float* mlp_w2 = (const float*)d_in[33], *mlp_b2 = (const float*)d_in[34];

    // ---- workspace layout ----
    char* p = (char*)d_ws;
    float* X    = (float*)carve(p, (size_t)NTOK * CC * 4);        // running activation
    f16* W16    = (f16*)  carve(p, (size_t)(8 * CC * CC + 2 * HID * CC) * 2);
    f16* LN16   = (f16*)  carve(p, (size_t)NTOK * CC * 2);
    f16* LNC16  = (f16*)  carve(p, (size_t)NTOK * CC * 2);
    f16* GQ     = (f16*)  carve(p, (size_t)NWL * CC * 2);         // gathered q windows
    f16* GK     = (f16*)  carve(p, (size_t)NWL * CC * 2);         // gathered kv windows
    f16* QP     = (f16*)  carve(p, (size_t)NWL * CC * 2);
    f16* KP     = (f16*)  carve(p, (size_t)NWL * CC * 2);
    f16* VP     = (f16*)  carve(p, (size_t)NWL * CC * 2);
    f16* OW     = (f16*)  carve(p, (size_t)NWL * CC * 2);
    float* YW   = (float*)carve(p, (size_t)NWL * CC * 4);
    f16* H1     = GQ;  // MLP hidden reuses gather buffers (59 MB < GQ..VP span)

    f16* Wsaq = W16 + 0 * CC * CC;
    f16* Wsak = W16 + 1 * CC * CC;
    f16* Wsav = W16 + 2 * CC * CC;
    f16* Wsao = W16 + 3 * CC * CC;
    f16* Wcaq = W16 + 4 * CC * CC;
    f16* Wcak = W16 + 5 * CC * CC;
    f16* Wcav = W16 + 6 * CC * CC;
    f16* Wcao = W16 + 7 * CC * CC;
    f16* Wm1  = W16 + 8 * CC * CC;
    f16* Wm2  = Wm1 + HID * CC;

    const int WSQ = CC * CC;            // 65536
    const int WM  = HID * CC;           // 262144
    // ---- convert weights to f16 ----
    f32_to_f16_kernel<<<WSQ / 256, 256, 0, stream>>>(sa_q_w, Wsaq, WSQ);
    f32_to_f16_kernel<<<WSQ / 256, 256, 0, stream>>>(sa_k_w, Wsak, WSQ);
    f32_to_f16_kernel<<<WSQ / 256, 256, 0, stream>>>(sa_v_w, Wsav, WSQ);
    f32_to_f16_kernel<<<WSQ / 256, 256, 0, stream>>>(sa_o_w, Wsao, WSQ);
    f32_to_f16_kernel<<<WSQ / 256, 256, 0, stream>>>(ca_q_w, Wcaq, WSQ);
    f32_to_f16_kernel<<<WSQ / 256, 256, 0, stream>>>(ca_k_w, Wcak, WSQ);
    f32_to_f16_kernel<<<WSQ / 256, 256, 0, stream>>>(ca_v_w, Wcav, WSQ);
    f32_to_f16_kernel<<<WSQ / 256, 256, 0, stream>>>(ca_o_w, Wcao, WSQ);
    f32_to_f16_kernel<<<WM / 256, 256, 0, stream>>>(mlp_w1, Wm1, WM);
    f32_to_f16_kernel<<<WM / 256, 256, 0, stream>>>(mlp_w2, Wm2, WM);

    const int NELT = NTOK * CC;
    copy_f32_kernel<<<NELT / 256, 256, 0, stream>>>(x_in, X, NELT);

    dim3 gProj(CC / 64, NWL / 128);       // (4, 256)
    dim3 attnGrid(NWIN, 2);

    // ================= self window attention =================
    ln_f16_kernel<<<NTOK / 8, 256, 0, stream>>>(X, ln1_w, ln1_b, LN16, NTOK);
    gather_windows_kernel<<<NWL, 256, 0, stream>>>(LN16, GQ);
    gemm_wmma_kernel<<<gProj, 256, 0, stream>>>(GQ, Wsaq, sa_q_b, QP, nullptr, NWL, CC, CC, 0);
    gemm_wmma_kernel<<<gProj, 256, 0, stream>>>(GQ, Wsak, sa_k_b, KP, nullptr, NWL, CC, CC, 0);
    gemm_wmma_kernel<<<gProj, 256, 0, stream>>>(GQ, Wsav, sa_v_b, VP, nullptr, NWL, CC, CC, 0);
    win_attn_kernel<<<attnGrid, 128, 0, stream>>>(QP, KP, VP, sa_bias, OW);
    gemm_wmma_kernel<<<gProj, 256, 0, stream>>>(OW, Wsao, sa_o_b, nullptr, YW, NWL, CC, CC, 0);
    scatter_residual_kernel<<<NTOK, 256, 0, stream>>>(X, YW, gamma1);

    // ================= cross window attention =================
    ln_f16_kernel<<<NTOK / 8, 256, 0, stream>>>(X, ln2_w, ln2_b, LN16, NTOK);
    ln_f16_kernel<<<NTOK / 8, 256, 0, stream>>>(ctx_in, lnc_w, lnc_b, LNC16, NTOK);
    gather_windows_kernel<<<NWL, 256, 0, stream>>>(LN16, GQ);
    gather_windows_kernel<<<NWL, 256, 0, stream>>>(LNC16, GK);
    gemm_wmma_kernel<<<gProj, 256, 0, stream>>>(GQ, Wcaq, ca_q_b, QP, nullptr, NWL, CC, CC, 0);
    gemm_wmma_kernel<<<gProj, 256, 0, stream>>>(GK, Wcak, ca_k_b, KP, nullptr, NWL, CC, CC, 0);
    gemm_wmma_kernel<<<gProj, 256, 0, stream>>>(GK, Wcav, ca_v_b, VP, nullptr, NWL, CC, CC, 0);
    win_attn_kernel<<<attnGrid, 128, 0, stream>>>(QP, KP, VP, ca_bias, OW);
    gemm_wmma_kernel<<<gProj, 256, 0, stream>>>(OW, Wcao, ca_o_b, nullptr, YW, NWL, CC, CC, 0);
    scatter_residual_kernel<<<NTOK, 256, 0, stream>>>(X, YW, gamma2);

    // ================= MLP =================
    ln_f16_kernel<<<NTOK / 8, 256, 0, stream>>>(X, ln3_w, ln3_b, LN16, NTOK);
    dim3 gMlp1(HID / 64, NTOK / 128);    // (16, 225)
    gemm_wmma_kernel<<<gMlp1, 256, 0, stream>>>(LN16, Wm1, mlp_b1, H1, nullptr,
                                                NTOK, HID, CC, /*gelu*/1);
    dim3 gMlp2(CC / 64, NTOK / 128);     // (4, 225)
    gemm_wmma_kernel<<<gMlp2, 256, 0, stream>>>(H1, Wm2, mlp_b2, nullptr, YW,
                                                NTOK, CC, HID, 0);
    final_add_kernel<<<NELT / 256, 256, 0, stream>>>(X, YW, gamma_mlp, (float*)d_out, NELT);
}
